// EBSDDI_67783173865717
// MI455X (gfx1250) — matrix operand order repaired and checked by reading, not compiled, not run on hardware
//
#include <hip/hip_runtime.h>
#include <stdint.h>

// Problem constants (from reference)
#define N_EXP  2048
#define N_DICT 65536
#define N_PIX  1600
#define TOPK   10

// GEMM tiling
#define BM 128          // query rows per block
#define BN 128          // dict cols per inner tile
#define KB 64           // K per LDS phase (2 WMMA K-substeps of 32)
#define CHUNK_N 1024    // dict cols handled per block
#define NT (CHUNK_N / BN)            // 8 inner tiles
#define NCHUNKS (N_DICT / CHUNK_N)   // 64

typedef __attribute__((ext_vector_type(16))) __bf16 v16bf;
typedef __attribute__((ext_vector_type(8)))  float  v8f;

union FragAB { v16bf bf; uint4 q[2]; };

// ---------------- helpers ----------------

__device__ __forceinline__ unsigned short f2bf(float f) {
  uint32_t u = __float_as_uint(f);
  u += 0x7FFFu + ((u >> 16) & 1u);   // round to nearest even
  return (unsigned short)(u >> 16);
}

// block = 256 threads = 8 waves (wave32)
__device__ __forceinline__ float blockReduceSum256(float v, float* red8) {
  #pragma unroll
  for (int o = 16; o > 0; o >>= 1) v += __shfl_xor(v, o, 32);
  const int wid  = threadIdx.x >> 5;
  const int lane = threadIdx.x & 31;
  if (lane == 0) red8[wid] = v;
  __syncthreads();
  float t = 0.f;
  #pragma unroll
  for (int w = 0; w < 8; ++w) t += red8[w];
  __syncthreads();   // allow red8 reuse
  return t;
}

// branchless sorted-descending top-k insert; static indices only (stays in VGPRs)
__device__ __forceinline__ void topkInsert(float v, int gidx, float (&tv)[TOPK], int (&ti)[TOPK]) {
  float cv = v; int ci = gidx;
  #pragma unroll
  for (int j = 0; j < TOPK; ++j) {
    const bool sw = cv > tv[j];
    const float t0 = tv[j]; const int i0 = ti[j];
    tv[j] = sw ? cv : t0;   ti[j] = sw ? ci : i0;
    cv    = sw ? t0 : cv;   ci    = sw ? i0 : ci;
  }
}

// ---------------- kernel 1/2: row center+normalize -> bf16 ----------------

__global__ void __launch_bounds__(256) rownorm_bf16_kernel(const float* __restrict__ in,
                                                           unsigned short* __restrict__ out) {
  __shared__ float red8[8];
  const size_t row = blockIdx.x;
  const float* __restrict__ src = in + row * (size_t)N_PIX;
  unsigned short* __restrict__ dst = out + row * (size_t)N_PIX;

  float vals[7];
  float s = 0.f;
  #pragma unroll
  for (int j = 0; j < 7; ++j) {
    const int i = threadIdx.x + j * 256;
    float x = 0.f;
    if (i < N_PIX) x = src[i];
    vals[j] = x;
    s += x;
  }
  const float mean = blockReduceSum256(s, red8) * (1.0f / (float)N_PIX);

  float ss = 0.f;
  #pragma unroll
  for (int j = 0; j < 7; ++j) {
    const int i = threadIdx.x + j * 256;
    if (i < N_PIX) { const float d = vals[j] - mean; ss += d * d; }
  }
  const float rn = rsqrtf(blockReduceSum256(ss, red8));

  #pragma unroll
  for (int j = 0; j < 7; ++j) {
    const int i = threadIdx.x + j * 256;
    if (i < N_PIX) dst[i] = f2bf((vals[j] - mean) * rn);
  }
}

// ---------------- kernel 3: bf16 WMMA GEMM + fused per-chunk top-k ----------------

// A/B staging tiles and the C spill tile are never live at the same time
// (barriers separate the K loop from the C write / top-k scan), so overlay
// them: total static LDS = 64 KB (5 concurrent blocks against 320 KB WGP pool).
union SharedTiles {
  struct { unsigned short a[BM * KB]; unsigned short b[BN * KB]; } ab;  // 16 KB + 16 KB
  float c[BM * BN];                                                     // 64 KB
};

// stage phase (tt, kk0) of A/B into 8 named uint4 registers (no arrays ->
// no allocas -> no scratch). bOff is in bf16 elements.
#define STAGE_GLOBAL(tt, kk0)                                                  \
  do {                                                                         \
    const size_t bOff = (size_t)(tt) * BN * N_PIX + (size_t)(kk0);             \
    tA0 = *(const uint4*)(aP0 + (kk0));                                        \
    tA1 = *(const uint4*)(aP1 + (kk0));                                        \
    tA2 = *(const uint4*)(aP2 + (kk0));                                        \
    tA3 = *(const uint4*)(aP3 + (kk0));                                        \
    tB0 = *(const uint4*)(bP + bOff);                                          \
    tB1 = *(const uint4*)(bP + bOff + (size_t)32 * N_PIX);                     \
    tB2 = *(const uint4*)(bP + bOff + (size_t)64 * N_PIX);                     \
    tB3 = *(const uint4*)(bP + bOff + (size_t)96 * N_PIX);                     \
    __builtin_prefetch((const void*)(bP + bOff + KB), 0, 1); /* warm L2 */     \
  } while (0)

// waves_per_eu(1): LDS (64 KB/block) bounds concurrency; give the register
// allocator the full VGPR file so the staging pipeline never spills.
__global__ void __launch_bounds__(256)
__attribute__((amdgpu_waves_per_eu(1)))
gemm_topk_kernel(
    const unsigned short* __restrict__ qn,   // [N_EXP][N_PIX] normalized bf16
    const unsigned short* __restrict__ dn,   // [N_DICT][N_PIX] normalized bf16
    float* __restrict__ candVal,             // [N_EXP][NCHUNKS][TOPK]
    int*   __restrict__ candIdx) {           // [N_EXP][NCHUNKS][TOPK]

  __shared__ SharedTiles sh;

  const int tid   = threadIdx.x;
  const int lane  = tid & 31;
  const int wave  = tid >> 5;
  const int waveM = (wave & 3) * 32;         // 4 waves stacked over M (128)
  const int waveN = (wave >> 2) * 64;        // 2 waves over N (128)
  const int kh    = lane >> 4;               // half-wave K-split selector
  const int l16   = lane & 15;

  const int qBase      = blockIdx.x * BM;
  const int chunk      = blockIdx.y;
  const int nChunkBase = chunk * CHUNK_N;

  // staging geometry: thread covers rows rowBase+32r, fixed 16B column slot
  const int rowBase = tid >> 3;              // 0..31
  const int colOff  = (tid & 7) * 8;         // bf16 elements (16B slots)

  // loop-invariant staging base pointers
  const unsigned short* aP0 = qn + (size_t)(qBase + rowBase +  0) * N_PIX + colOff;
  const unsigned short* aP1 = qn + (size_t)(qBase + rowBase + 32) * N_PIX + colOff;
  const unsigned short* aP2 = qn + (size_t)(qBase + rowBase + 64) * N_PIX + colOff;
  const unsigned short* aP3 = qn + (size_t)(qBase + rowBase + 96) * N_PIX + colOff;
  const unsigned short* bP  = dn + (size_t)(nChunkBase + rowBase) * N_PIX + colOff;

  // LDS commit slots (row stride KB elements; +32 rows per slab)
  unsigned short* ldsAslot = sh.ab.a + rowBase * KB + colOff;
  unsigned short* ldsBslot = sh.ab.b + rowBase * KB + colOff;

  float tv[TOPK]; int ti[TOPK];
  #pragma unroll
  for (int j = 0; j < TOPK; ++j) { tv[j] = -2.0f; ti[j] = 0; }

  // register staging (software pipeline: load p+1 while computing p)
  uint4 tA0, tA1, tA2, tA3, tB0, tB1, tB2, tB3;
  STAGE_GLOBAL(0, 0);   // prologue: phase 0 in flight

  for (int t = 0; t < NT; ++t) {
    const int nBase = nChunkBase + t * BN;
    v8f acc[2][4] = {};

    for (int k0 = 0; k0 < N_PIX; k0 += KB) {
      __syncthreads();
      // commit staged registers to LDS
      *(uint4*)(ldsAslot + 0 * 32 * KB) = tA0;
      *(uint4*)(ldsAslot + 1 * 32 * KB) = tA1;
      *(uint4*)(ldsAslot + 2 * 32 * KB) = tA2;
      *(uint4*)(ldsAslot + 3 * 32 * KB) = tA3;
      *(uint4*)(ldsBslot + 0 * 32 * KB) = tB0;
      *(uint4*)(ldsBslot + 1 * 32 * KB) = tB1;
      *(uint4*)(ldsBslot + 2 * 32 * KB) = tB2;
      *(uint4*)(ldsBslot + 3 * 32 * KB) = tB3;
      __syncthreads();

      // kick off next phase's global loads (uniform branch)
      {
        int nk = k0 + KB, ntile = t;
        if (nk == N_PIX) { nk = 0; ++ntile; }
        if (ntile < NT) STAGE_GLOBAL(ntile, nk);
      }

      // --- two WMMA K-substeps per LDS phase; fragments via 128-bit DS loads ---
      #pragma unroll
      for (int ks = 0; ks < 2; ++ks) {
        const int kk = ks * 32;
        FragAB a[2], b[4];
        #pragma unroll
        for (int tm = 0; tm < 2; ++tm) {
          const unsigned short* base = sh.ab.a + (waveM + tm * 16 + l16) * KB + kk;
          a[tm].q[0] = *(const uint4*)(base + kh * 8);        // K = 8h .. 8h+7
          a[tm].q[1] = *(const uint4*)(base + 16 + kh * 8);   // K = 16+8h .. +7
        }
        #pragma unroll
        for (int tn = 0; tn < 4; ++tn) {
          const unsigned short* base =
              sh.ab.b + (waveN + tn * 16 + l16) * KB + kk;
          b[tn].q[0] = *(const uint4*)(base + kh * 8);        // K = 8h .. 8h+7
          b[tn].q[1] = *(const uint4*)(base + 16 + kh * 8);   // K = 16+8h .. +7
        }
        #pragma unroll
        for (int tm = 0; tm < 2; ++tm)
          #pragma unroll
          for (int tn = 0; tn < 4; ++tn)
            acc[tm][tn] = __builtin_amdgcn_wmma_f32_16x16x32_bf16(
                false, a[tm].bf, false, b[tn].bf, (short)0, acc[tm][tn], false, false);
      }
    }

    // --- spill C tile to LDS (overlays A/B staging), then per-row top-k scan ---
    __syncthreads();
    #pragma unroll
    for (int tm = 0; tm < 2; ++tm)
      #pragma unroll
      for (int tn = 0; tn < 4; ++tn)
        #pragma unroll
        for (int v = 0; v < 8; ++v)
          sh.c[(waveM + tm * 16 + kh * 8 + v) * BN + (waveN + tn * 16 + l16)] = acc[tm][tn][v];
    __syncthreads();

    if (tid < BM) {
      const float* rowC = sh.c + tid * BN;
      #pragma unroll 4
      for (int n = 0; n < BN; ++n) {
        const float v = rowC[n];
        if (v > tv[TOPK - 1]) topkInsert(v, nBase + n, tv, ti);
      }
    }
    // next iteration's leading __syncthreads() protects LDS reuse
  }

  if (tid < BM) {
    const size_t base = ((size_t)(qBase + tid) * NCHUNKS + chunk) * TOPK;
    #pragma unroll
    for (int j = 0; j < TOPK; ++j) { candVal[base + j] = tv[j]; candIdx[base + j] = ti[j]; }
  }
}

// ---------------- kernel 4: merge candidates, arccos, gather quaternions ----------------

__global__ void __launch_bounds__(256) merge_topk_kernel(
    const float* __restrict__ candVal,
    const int*   __restrict__ candIdx,
    const float* __restrict__ so3,      // [N_DICT][4]
    float* __restrict__ out) {
  const int q = blockIdx.x * 256 + threadIdx.x;
  if (q >= N_EXP) return;

  float tv[TOPK]; int ti[TOPK];
  #pragma unroll
  for (int j = 0; j < TOPK; ++j) { tv[j] = -2.0f; ti[j] = 0; }

  const float* cv = candVal + (size_t)q * NCHUNKS * TOPK;
  const int*   ci = candIdx + (size_t)q * NCHUNKS * TOPK;
  for (int k = 0; k < NCHUNKS * TOPK; ++k) {
    const float v = cv[k];
    if (v > tv[TOPK - 1]) topkInsert(v, ci[k], tv, ti);
  }

  float* outAng = out;                                   // [N_EXP][TOPK]
  float* outIdx = out + (size_t)N_EXP * TOPK;            // [N_EXP][TOPK] (as float)
  float* outOri = out + (size_t)N_EXP * TOPK * 2;        // [N_EXP][TOPK][4]
  #pragma unroll
  for (int j = 0; j < TOPK; ++j) {
    const float c = fminf(1.0f, fmaxf(-1.0f, tv[j]));
    outAng[(size_t)q * TOPK + j] = acosf(c);
    outIdx[(size_t)q * TOPK + j] = (float)ti[j];
    #pragma unroll
    for (int c4 = 0; c4 < 4; ++c4)
      outOri[((size_t)q * TOPK + j) * 4 + c4] = so3[(size_t)ti[j] * 4 + c4];
  }
}

// ---------------- launch ----------------

extern "C" void kernel_launch(void* const* d_in, const int* in_sizes, int n_in,
                              void* d_out, int out_size, void* d_ws, size_t ws_size,
                              hipStream_t stream) {
  const float* expd = (const float*)d_in[0];   // [N_EXP][N_PIX]
  const float* pats = (const float*)d_in[1];   // [N_DICT][N_PIX]
  const float* so3  = (const float*)d_in[2];   // [N_DICT][4]
  (void)in_sizes; (void)n_in; (void)out_size; (void)ws_size;  // topk fixed at 10

  // workspace layout (bytes, all 256B-aligned offsets)
  char* ws = (char*)d_ws;
  unsigned short* qn = (unsigned short*)(ws);                       //   6,553,600 B
  unsigned short* dn = (unsigned short*)(ws + 6553600);             // 209,715,200 B
  float* candVal     = (float*)(ws + 6553600 + 209715200);          //   5,242,880 B
  int*   candIdx     = (int*)  (ws + 6553600 + 209715200 + 5242880);//   5,242,880 B

  hipLaunchKernelGGL(rownorm_bf16_kernel, dim3(N_EXP),  dim3(256), 0, stream, expd, qn);
  hipLaunchKernelGGL(rownorm_bf16_kernel, dim3(N_DICT), dim3(256), 0, stream, pats, dn);
  hipLaunchKernelGGL(gemm_topk_kernel, dim3(N_EXP / BM, NCHUNKS), dim3(256), 0, stream,
                     qn, dn, candVal, candIdx);
  hipLaunchKernelGGL(merge_topk_kernel, dim3((N_EXP + 255) / 256), dim3(256), 0, stream,
                     candVal, candIdx, so3, (float*)d_out);
}
